// SurgeGNN_10282151707180
// MI455X (gfx1250) — matrix operand order repaired and checked by reading, not compiled, not run on hardware
//
#include <hip/hip_runtime.h>
#include <math.h>

typedef _Float16 half8 __attribute__((ext_vector_type(8)));
typedef _Float16 v16h  __attribute__((ext_vector_type(16)));
typedef float    v8f   __attribute__((ext_vector_type(8)));

// ---------------------------------------------------------------- utilities

__global__ void k_zero_f32(float* __restrict__ p, int count) {
    int i = blockIdx.x * 256 + threadIdx.x;
    if (i < count) p[i] = 0.0f;
}

__global__ void k_count(const int* __restrict__ dst, float* __restrict__ cnt, int E) {
    int e = blockIdx.x * 256 + threadIdx.x;
    if (e < E) atomicAdd(&cnt[dst[e]], 1.0f);
}

__global__ void k_invert(float* __restrict__ cnt, int n) {
    int i = blockIdx.x * 256 + threadIdx.x;
    if (i < n) cnt[i] = 1.0f / fmaxf(cnt[i], 1.0f);
}

// edge-parallel scatter-add of 4-float chunks: msum[dst] += x[src]
__global__ void k_scatter4(const float* __restrict__ x, const int* __restrict__ src,
                           const int* __restrict__ dst, float* __restrict__ msum,
                           int E, int F, int csh /* log2(F/4) */) {
    int tid = blockIdx.x * 256 + threadIdx.x;
    int chunks = 1 << csh;
    if (tid >= E * chunks) return;
    int e = tid >> csh;
    int c = (tid & (chunks - 1)) << 2;
    int s = src[e], d = dst[e];
    const float4 v = *(const float4*)(x + (size_t)s * F + c);
    float* o = msum + (size_t)d * F + c;
    atomicAdd(o + 0, v.x);
    atomicAdd(o + 1, v.y);
    atomicAdd(o + 2, v.z);
    atomicAdd(o + 3, v.w);
}

// ---------------------------------------------------------------- weight prep

// Bc[fo][k] (f16, k in [0,2*Fi)): k<Fi -> w_l[fo][k] else w_r[fo][k-Fi]
__global__ void k_w_concat_h(const float* __restrict__ wl, const float* __restrict__ wr,
                             _Float16* __restrict__ Bc, int Fo, int Fi) {
    int K = 2 * Fi;
    int tid = blockIdx.x * 256 + threadIdx.x;
    if (tid >= Fo * K) return;
    int fo = tid / K;
    int k  = tid - fo * K;
    float v = (k < Fi) ? wl[(size_t)fo * Fi + k] : wr[(size_t)fo * Fi + (k - Fi)];
    Bc[tid] = (_Float16)v;
}

__global__ void k_w_h(const float* __restrict__ w, _Float16* __restrict__ Bc, int count) {
    int tid = blockIdx.x * 256 + threadIdx.x;
    if (tid < count) Bc[tid] = (_Float16)w[tid];
}

// fold bias + batchnorm: out = mm*scale + shift ; scale=g*rsqrt(rv+eps), shift=(bl-rm)*scale+be
__global__ void k_bnfold(const float* __restrict__ bl, const float* __restrict__ g,
                         const float* __restrict__ be, const float* __restrict__ rm,
                         const float* __restrict__ rv,
                         float* __restrict__ scale, float* __restrict__ shift, int Fo) {
    int i = blockIdx.x * 256 + threadIdx.x;
    if (i >= Fo) return;
    float s = g[i] * rsqrtf(rv[i] + 1e-5f);
    scale[i] = s;
    shift[i] = (bl[i] - rm[i]) * s + be[i];
}

__global__ void k_biasfold(const float* __restrict__ pb,
                           float* __restrict__ scale, float* __restrict__ shift, int Fo) {
    int i = blockIdx.x * 256 + threadIdx.x;
    if (i >= Fo) return;
    scale[i] = 1.0f;
    shift[i] = pb[i];
}

// ---------------------------------------------------------------- fused WMMA GEMM
// out[n x Fo] = relu( ([a0*inv | a1])[n x K] @ Bc[Fo x K]^T * scale + shift )
// a0 row-stride = K0 ; a1 row-stride = K-K0 (a1 may be null when K0==K)
// K in {256,512} (multiple of 32), Fo multiple of 16, 16 rows per block, 8 waves.
__global__ __launch_bounds__(256)
void k_wmma_gemm(const float* __restrict__ a0, const float* __restrict__ inv,
                 const float* __restrict__ a1, const _Float16* __restrict__ Bc,
                 const float* __restrict__ scale, const float* __restrict__ shift,
                 float* __restrict__ out, int n, int K0, int K, int Fo) {
    __shared__ __align__(32) _Float16 Ash[16][512 + 8];   // padded: 520 halves/row
    const int nBase = blockIdx.x * 16;
    const int ksh = (K == 512) ? 9 : 8;

    // stage 16 x K A-panel to LDS as f16 (neighbor-mean part scaled by inv)
    for (int idx = threadIdx.x; idx < (16 << ksh); idx += 256) {
        int r = idx >> ksh;
        int k = idx & (K - 1);
        int row = nBase + r;
        if (row >= n) row = n - 1;
        float v;
        if (k < K0) {
            float iv = inv ? inv[row] : 1.0f;
            v = a0[(size_t)row * K0 + k] * iv;
        } else {
            v = a1[(size_t)row * (K - K0) + (k - K0)];
        }
        Ash[r][k] = (_Float16)v;
    }
    __syncthreads();

    const int wid  = threadIdx.x >> 5;
    const int lane = threadIdx.x & 31;
    const int hlf  = lane >> 4;            // half-wave select
    const int arow = lane & 15;            // A row (M) / B column within tile
    const int aoff = hlf << 3;             // A: 0 or 8 halves
    const int boff = hlf << 3;             // B: 0 or 8 halves
    const int nTiles = Fo >> 4;

    for (int t = wid; t < nTiles; t += 8) {
        const int cbase = t << 4;
        const int col = cbase + arow;
        const _Float16* brow = Bc + (size_t)col * K;
        v8f acc = {};
        for (int kb = 0; kb < K; kb += 32) {
            // A fragment: lane<16 holds M=lane, K {kb..kb+7, kb+16..kb+23};
            //             lane>=16 holds K {kb+8..15, kb+24..31}
            const _Float16* ap = &Ash[arow][kb + aoff];
            half8 alo = *(const half8*)(ap);
            half8 ahi = *(const half8*)(ap + 16);
            v16h a;
#pragma unroll
            for (int i = 0; i < 8; ++i) { a[i] = alo[i]; a[i + 8] = ahi[i]; }
            // B fragment: lane<16 -> K {kb..kb+7, kb+16..kb+23}, lane>=16 -> K {kb+8..15, kb+24..31}, col N=arow
            half8 blo = *(const half8*)(brow + kb + boff);
            half8 bhi = *(const half8*)(brow + kb + boff + 16);
            v16h b;
#pragma unroll
            for (int i = 0; i < 8; ++i) { b[i] = blo[i]; b[i + 8] = bhi[i]; }
            acc = __builtin_amdgcn_wmma_f32_16x16x32_f16(
                false, a, false, b, (short)0, acc, false, false);
        }
        // C/D layout: VGPR r -> M=r (lanes 0-15) or M=r+8 (lanes 16-31), N=arow
        float s  = scale[col];
        float sh = shift[col];
        int rbase = nBase + (hlf << 3);
#pragma unroll
        for (int r = 0; r < 8; ++r) {
            int row = rbase + r;
            if (row < n) {
                float v = acc[r] * s + sh;
                out[(size_t)row * Fo + col] = v > 0.0f ? v : 0.0f;
            }
        }
    }
}

// ---------------------------------------------------------------- final head: [N x K] -> sigmoid dot
__global__ void k_final(const float* __restrict__ h, const float* __restrict__ w,
                        const float* __restrict__ b, float* __restrict__ out, int n, int K) {
    int i = blockIdx.x * 256 + threadIdx.x;
    if (i >= n) return;
    const float4* hp = (const float4*)(h + (size_t)i * K);
    const float4* wp = (const float4*)w;
    float sum = 0.0f;
    for (int k = 0; k < (K >> 2); ++k) {
        float4 a = hp[k], c = wp[k];
        sum += a.x * c.x + a.y * c.y + a.z * c.z + a.w * c.w;
    }
    sum += b[0];
    out[i] = 1.0f / (1.0f + expf(-sum));
}

// ---------------------------------------------------------------- launch

extern "C" void kernel_launch(void* const* d_in, const int* in_sizes, int n_in,
                              void* d_out, int out_size, void* d_ws, size_t ws_size,
                              hipStream_t stream) {
    const int IN = 128, H = 256;
    const float* x  = (const float*)d_in[0];
    const int*   ei = (const int*)d_in[1];
    const int n = in_sizes[0] / IN;
    const int E = in_sizes[1] / 2;
    const int* src = ei;
    const int* dst = ei + E;

    const float* wl[3] = {(const float*)d_in[2],  (const float*)d_in[9],  (const float*)d_in[16]};
    const float* bl[3] = {(const float*)d_in[3],  (const float*)d_in[10], (const float*)d_in[17]};
    const float* wr[3] = {(const float*)d_in[4],  (const float*)d_in[11], (const float*)d_in[18]};
    const float* gg[3] = {(const float*)d_in[5],  (const float*)d_in[12], (const float*)d_in[19]};
    const float* bb[3] = {(const float*)d_in[6],  (const float*)d_in[13], (const float*)d_in[20]};
    const float* rm[3] = {(const float*)d_in[7],  (const float*)d_in[14], (const float*)d_in[21]};
    const float* rv[3] = {(const float*)d_in[8],  (const float*)d_in[15], (const float*)d_in[22]};
    const float* pw1 = (const float*)d_in[23]; const float* pb1 = (const float*)d_in[24];
    const float* pw2 = (const float*)d_in[25]; const float* pb2 = (const float*)d_in[26];
    const float* pw3 = (const float*)d_in[27]; const float* pb3 = (const float*)d_in[28];

    // workspace carve-out (256B aligned slots)
    char* ws = (char*)d_ws;
    size_t off = 0;
    auto take = [&](size_t bytes) -> void* {
        void* p = ws + off;
        off += (bytes + 255) & ~(size_t)255;
        return p;
    };
    float*    inv  = (float*)take((size_t)n * 4);
    float*    P    = (float*)take((size_t)n * H * 4);
    float*    Q    = (float*)take((size_t)n * H * 4);
    _Float16* w0c  = (_Float16*)take((size_t)H * 2 * IN * 2);   // 256 x 256 halves
    _Float16* w1c  = (_Float16*)take((size_t)H * 2 * H * 2);    // 256 x 512
    _Float16* w2c  = (_Float16*)take((size_t)H * 2 * H * 2);    // 256 x 512
    _Float16* wp1c = (_Float16*)take((size_t)H * H * 2);        // 256 x 256
    _Float16* wp2c = (_Float16*)take((size_t)(H / 2) * H * 2);  // 128 x 256
    float* sc0 = (float*)take(H * 4); float* sh0 = (float*)take(H * 4);
    float* sc1 = (float*)take(H * 4); float* sh1 = (float*)take(H * 4);
    float* sc2 = (float*)take(H * 4); float* sh2 = (float*)take(H * 4);
    float* scp1 = (float*)take(H * 4); float* shp1 = (float*)take(H * 4);
    float* scp2 = (float*)take((H / 2) * 4); float* shp2 = (float*)take((H / 2) * 4);
    (void)ws_size; (void)n_in;

    auto blocks = [](long count) { return (unsigned)((count + 255) / 256); };
    const unsigned gemmGrid = (unsigned)((n + 15) / 16);

    // degree counts -> inverse
    k_zero_f32<<<blocks(n), 256, 0, stream>>>(inv, n);
    k_count  <<<blocks(E), 256, 0, stream>>>(dst, inv, E);
    k_invert <<<blocks(n), 256, 0, stream>>>(inv, n);

    // weight conversion + BN/bias folding
    k_w_concat_h<<<blocks((long)H * 2 * IN), 256, 0, stream>>>(wl[0], wr[0], w0c, H, IN);
    k_w_concat_h<<<blocks((long)H * 2 * H),  256, 0, stream>>>(wl[1], wr[1], w1c, H, H);
    k_w_concat_h<<<blocks((long)H * 2 * H),  256, 0, stream>>>(wl[2], wr[2], w2c, H, H);
    k_w_h<<<blocks((long)H * H),       256, 0, stream>>>(pw1, wp1c, H * H);
    k_w_h<<<blocks((long)(H / 2) * H), 256, 0, stream>>>(pw2, wp2c, (H / 2) * H);
    k_bnfold<<<1, 256, 0, stream>>>(bl[0], gg[0], bb[0], rm[0], rv[0], sc0, sh0, H);
    k_bnfold<<<1, 256, 0, stream>>>(bl[1], gg[1], bb[1], rm[1], rv[1], sc1, sh1, H);
    k_bnfold<<<1, 256, 0, stream>>>(bl[2], gg[2], bb[2], rm[2], rv[2], sc2, sh2, H);
    k_biasfold<<<1, 256, 0, stream>>>(pb1, scp1, shp1, H);
    k_biasfold<<<1, 256, 0, stream>>>(pb2, scp2, shp2, H / 2);

    // ---- layer 0: aggregate x (F=128) into P, gemm([P*inv | x] @ W0) -> Q
    k_zero_f32<<<blocks((long)n * IN), 256, 0, stream>>>(P, n * IN);
    k_scatter4<<<blocks((long)E * (IN / 4)), 256, 0, stream>>>(x, src, dst, P, E, IN, 5);
    k_wmma_gemm<<<gemmGrid, 256, 0, stream>>>(P, inv, x, w0c, sc0, sh0, Q, n, IN, IN + IN, H);

    // ---- layer 1: aggregate Q (F=256) into P, gemm([P*inv | Q] @ W1) -> P (in-place safe)
    k_zero_f32<<<blocks((long)n * H), 256, 0, stream>>>(P, n * H);
    k_scatter4<<<blocks((long)E * (H / 4)), 256, 0, stream>>>(Q, src, dst, P, E, H, 6);
    k_wmma_gemm<<<gemmGrid, 256, 0, stream>>>(P, inv, Q, w1c, sc1, sh1, P, n, H, 2 * H, H);

    // ---- layer 2: aggregate P into Q, gemm([Q*inv | P] @ W2) -> Q
    k_zero_f32<<<blocks((long)n * H), 256, 0, stream>>>(Q, n * H);
    k_scatter4<<<blocks((long)E * (H / 4)), 256, 0, stream>>>(P, src, dst, Q, E, H, 6);
    k_wmma_gemm<<<gemmGrid, 256, 0, stream>>>(Q, inv, P, w2c, sc2, sh2, Q, n, H, 2 * H, H);

    // ---- predictor MLP
    k_wmma_gemm<<<gemmGrid, 256, 0, stream>>>(Q, nullptr, nullptr, wp1c, scp1, shp1, P, n, H, H, H);
    k_wmma_gemm<<<gemmGrid, 256, 0, stream>>>(P, nullptr, nullptr, wp2c, scp2, shp2, Q, n, H, H, H / 2);
    k_final<<<blocks(n), 256, 0, stream>>>(Q, pw3, pb3, (float*)d_out, n, H / 2);
}
